// PocketGNNv5_67705864454301
// MI455X (gfx1250) — matrix operand order repaired and checked by reading, compiled, not run on hardware
//
#include <hip/hip_runtime.h>

// ---------------------------------------------------------------------------
// Types for CDNA5 WMMA (wave32): V_WMMA_F32_16X16X32_BF16
// ---------------------------------------------------------------------------
typedef __attribute__((ext_vector_type(16))) __bf16 v16bf;
typedef __attribute__((ext_vector_type(8)))  float  v8f;
typedef __attribute__((ext_vector_type(4)))  unsigned int u32x4;

union Frag16 {
  u32x4 u[2];
  v16bf v;
};

__device__ __forceinline__ unsigned short f2bf(float f) {
  unsigned int u = __float_as_uint(f);
  u += 0x7FFFu + ((u >> 16) & 1u);   // round-to-nearest-even
  return (unsigned short)(u >> 16);
}

// float atomic max via int/uint atomics (valid with -inf init)
__device__ __forceinline__ void atomic_max_f32(float* a, float v) {
  if (v >= 0.0f) atomicMax((int*)a, __float_as_int(v));
  else           atomicMin((unsigned int*)a, __float_as_uint(v));
}

// A-fragment from LDS: lane = M row; two 8-element K chunks (16B each)
template<int STRIDE>
__device__ __forceinline__ v16bf lds_a_frag(const unsigned short (*A)[STRIDE],
                                            int lane, int k0) {
  Frag16 f;
  const int row  = lane & 15;
  const int koff = k0 + ((lane >> 4) << 3);
  f.u[0] = *(const u32x4*)&A[row][koff];
  f.u[1] = *(const u32x4*)&A[row][koff + 16];
  return f.v;
}

// B-fragment from global (pre-packed): 32 contiguous bytes per lane
__device__ __forceinline__ v16bf glb_b_frag(const unsigned short* __restrict__ Wp,
                                            int nt, int ks, int ksteps, int lane) {
  const unsigned short* p = Wp + ((((nt * ksteps) + ks) * 32 + lane) << 4);
  Frag16 f;
  f.u[0] = ((const u32x4*)p)[0];
  f.u[1] = ((const u32x4*)p)[1];
  return f.v;
}

// ---------------------------------------------------------------------------
// Utility kernels
// ---------------------------------------------------------------------------
__global__ void fill_f32(float* p, float v, int n) {
  int i = blockIdx.x * blockDim.x + threadIdx.x;
  if (i < n) p[i] = v;
}

__global__ void fill_u32(unsigned int* p, unsigned int v, int n) {
  int i = blockIdx.x * blockDim.x + threadIdx.x;
  if (i < n) p[i] = v;
}

__global__ void finalize_max(float* p, int n) {
  int i = blockIdx.x * blockDim.x + threadIdx.x;
  if (i < n) {
    float v = p[i];
    if (!(v >= -3.0e38f)) p[i] = 0.0f;   // -inf (no in-edge) or NaN -> 0
  }
}

// Pack fp32 weight [Ksrc x 256] (row-major) into bf16 B-fragment-major layout:
// idx = (((nt*KSTEPS + ks)*32 + lane)*16 + e)
// value = W[ks*32 + 8*(lane>>4) + (e&7) + 16*(e>>3)][nt*16 + (lane&15)], zero-padded K >= Ksrc
__global__ void pack_w(const float* __restrict__ W, unsigned short* __restrict__ Wp,
                       int Ksrc, int Kp) {
  int idx = blockIdx.x * 256 + threadIdx.x;
  int total = Kp * 256;
  if (idx >= total) return;
  int e      = idx & 15;
  int lane   = (idx >> 4) & 31;
  int ksteps = Kp >> 5;
  int t      = idx >> 9;
  int ks     = t % ksteps;
  int nt     = t / ksteps;
  int k = (ks << 5) + ((lane >> 4) << 3) + (e & 7) + ((e >> 3) << 4);
  int n = (nt << 4) + (lane & 15);
  float v = (k < Ksrc) ? W[k * 256 + n] : 0.0f;
  Wp[idx] = f2bf(v);
}

// ---------------------------------------------------------------------------
// EdgeConv: per block, 16 edges x (relu([xi|xj-xi]@W1+b1)@W2+b2), max-agg @dst
// 256 threads = 8 waves; each wave owns 2 N-tiles of 16 columns (8*2*16=256).
// K loops are software-pipelined one step to hide L2 latency behind WMMAs.
// ---------------------------------------------------------------------------
template<int KP, int DIN, int XS>
__global__ __launch_bounds__(256) void edge_conv(
    const float* __restrict__ x,
    const int* __restrict__ srcIdx, const int* __restrict__ dstIdx,
    const unsigned short* __restrict__ W1p, const float* __restrict__ b1,
    const unsigned short* __restrict__ W2p, const float* __restrict__ b2,
    float* __restrict__ agg, int nEdges)
{
  constexpr int AS_STRIDE = KP + 8;     // ushort units; 16B-aligned row stride
  constexpr int H_STRIDE  = 256 + 8;
  constexpr int KSTEPS1   = KP >> 5;
  constexpr int KSTEPS2   = 256 >> 5;
  constexpr int KV        = KP >> 2;    // float4 chunks per staged row

  __shared__ __align__(16) unsigned short As[16][AS_STRIDE];
  __shared__ __align__(16) unsigned short H1[16][H_STRIDE];
  __shared__ int eS[16], eD[16];

  const int tid  = threadIdx.x;
  const int lane = tid & 31;
  const int wave = tid >> 5;

  // prefetch second-stage weights into cache hierarchy
  __builtin_prefetch(W2p + (wave << 8), 0, 0);

  if (tid < 16) {
    int e = blockIdx.x * 16 + tid;
    if (e >= nEdges) e = nEdges - 1;     // duplicate edges are harmless for max
    eS[tid] = srcIdx[e];
    eD[tid] = dstIdx[e];
  }
  __syncthreads();

  // Stage A = [xi | xj - xi] as bf16 (float4 loads, 8B LDS stores), zero-padded
  for (int idx = tid; idx < 16 * KV; idx += 256) {
    int e  = idx / KV;                   // KV is a power of two -> shift
    int k4 = (idx - e * KV) << 2;
    int dn = eD[e], sn = eS[e];
    float v0 = 0.0f, v1 = 0.0f, v2 = 0.0f, v3 = 0.0f;
    if (k4 < DIN) {
      const float4 a = *(const float4*)&x[dn * XS + k4];
      v0 = a.x; v1 = a.y; v2 = a.z; v3 = a.w;
    } else if (k4 < 2 * DIN) {
      int kk = k4 - DIN;
      const float4 a = *(const float4*)&x[sn * XS + kk];
      const float4 b = *(const float4*)&x[dn * XS + kk];
      v0 = a.x - b.x; v1 = a.y - b.y; v2 = a.z - b.z; v3 = a.w - b.w;
    }
    unsigned int lo = (unsigned int)f2bf(v0) | ((unsigned int)f2bf(v1) << 16);
    unsigned int hi = (unsigned int)f2bf(v2) | ((unsigned int)f2bf(v3) << 16);
    *(uint2*)&As[e][k4] = make_uint2(lo, hi);
  }
  __syncthreads();

  const int nt0 = wave * 2;
  const int nt1 = nt0 + 1;

  // ---- Matmul 1: (16 x KP) @ (KP x 256) -> 16 x 256 (pipelined) ----
  v8f c0 = {}, c1 = {};
  {
    v16bf fa  = lds_a_frag(As, lane, 0);
    v16bf fb0 = glb_b_frag(W1p, nt0, 0, KSTEPS1, lane);
    v16bf fb1 = glb_b_frag(W1p, nt1, 0, KSTEPS1, lane);
#pragma unroll
    for (int ks = 0; ks < KSTEPS1; ++ks) {
      v16bf na = fa, nb0 = fb0, nb1 = fb1;
      if (ks + 1 < KSTEPS1) {
        na  = lds_a_frag(As, lane, (ks + 1) << 5);
        nb0 = glb_b_frag(W1p, nt0, ks + 1, KSTEPS1, lane);
        nb1 = glb_b_frag(W1p, nt1, ks + 1, KSTEPS1, lane);
      }
      c0 = __builtin_amdgcn_wmma_f32_16x16x32_bf16(false, fa, false, fb0,
                                                   (short)0, c0, false, false);
      c1 = __builtin_amdgcn_wmma_f32_16x16x32_bf16(false, fa, false, fb1,
                                                   (short)0, c1, false, false);
      fa = na; fb0 = nb0; fb1 = nb1;
    }
  }

  // bias + relu, write h1 (bf16) back to LDS as A-matrix for matmul 2
  {
    const int half = lane >> 4;
    const int col  = lane & 15;
    const int n0 = nt0 * 16 + col, n1 = nt1 * 16 + col;
    const float bb0 = b1[n0], bb1 = b1[n1];
#pragma unroll
    for (int v = 0; v < 8; ++v) {
      int r = v + half * 8;              // C/D layout: VGPR v -> row v (+8 hi half)
      H1[r][n0] = f2bf(fmaxf(c0[v] + bb0, 0.0f));
      H1[r][n1] = f2bf(fmaxf(c1[v] + bb1, 0.0f));
    }
  }
  __syncthreads();

  // ---- Matmul 2: (16 x 256) @ (256 x 256) -> 16 x 256 (pipelined) ----
  v8f d0 = {}, d1 = {};
  {
    v16bf fa  = lds_a_frag(H1, lane, 0);
    v16bf fb0 = glb_b_frag(W2p, nt0, 0, KSTEPS2, lane);
    v16bf fb1 = glb_b_frag(W2p, nt1, 0, KSTEPS2, lane);
#pragma unroll
    for (int ks = 0; ks < KSTEPS2; ++ks) {
      v16bf na = fa, nb0 = fb0, nb1 = fb1;
      if (ks + 1 < KSTEPS2) {
        na  = lds_a_frag(H1, lane, (ks + 1) << 5);
        nb0 = glb_b_frag(W2p, nt0, ks + 1, KSTEPS2, lane);
        nb1 = glb_b_frag(W2p, nt1, ks + 1, KSTEPS2, lane);
      }
      d0 = __builtin_amdgcn_wmma_f32_16x16x32_bf16(false, fa, false, fb0,
                                                   (short)0, d0, false, false);
      d1 = __builtin_amdgcn_wmma_f32_16x16x32_bf16(false, fa, false, fb1,
                                                   (short)0, d1, false, false);
      fa = na; fb0 = nb0; fb1 = nb1;
    }
  }

  // bias + segment-max into agg[dst]
  {
    const int half = lane >> 4;
    const int col  = lane & 15;
    const int n0 = nt0 * 16 + col, n1 = nt1 * 16 + col;
    const float bb0 = b2[n0], bb1 = b2[n1];
#pragma unroll
    for (int v = 0; v < 8; ++v) {
      int r = v + half * 8;
      int node = eD[r];
      atomic_max_f32(&agg[node * 256 + n0], d0[v] + bb0);
      atomic_max_f32(&agg[node * 256 + n1], d1[v] + bb1);
    }
  }
}

// ---------------------------------------------------------------------------
// Pooling + head
// ---------------------------------------------------------------------------
__global__ void pool_count(const int* __restrict__ batch, unsigned int* cnt, int n) {
  int i = blockIdx.x * blockDim.x + threadIdx.x;
  if (i < n) atomicAdd(&cnt[batch[i]], 1u);
}

__global__ void pool_accum(const float* __restrict__ xf, const int* __restrict__ batch,
                           float* __restrict__ gsum, float* __restrict__ gmax, int n) {
  int i = blockIdx.x * blockDim.x + threadIdx.x;
  if (i >= n) return;
  int node = i >> 8, c = i & 255;
  int g = batch[node];
  float v = xf[i];
  atomicAdd(&gsum[g * 256 + c], v);
  atomic_max_f32(&gmax[g * 256 + c], v);
}

__global__ __launch_bounds__(256) void head_mlp(
    const float* __restrict__ gsum, const float* __restrict__ gmax,
    const unsigned int* __restrict__ gcnt,
    const float* __restrict__ wc1, const float* __restrict__ bc1,
    const float* __restrict__ wc2, const float* __restrict__ bc2,
    const float* __restrict__ wc3, const float* __restrict__ bc3,
    float* __restrict__ out)
{
  __shared__ float gv[512];
  __shared__ float h1s[256];
  __shared__ float h2s[64];
  int g = blockIdx.x, t = threadIdx.x;
  float cnt = (float)gcnt[g];
  if (cnt < 1.0f) cnt = 1.0f;
  {
    gv[t] = gsum[g * 256 + t] / cnt;
    float m = gmax[g * 256 + t];
    if (!(m >= -3.0e38f)) m = 0.0f;
    gv[256 + t] = m;
  }
  __syncthreads();
  float a = bc1[t];
  for (int k = 0; k < 512; ++k) a += gv[k] * wc1[k * 256 + t];
  h1s[t] = fmaxf(a, 0.0f);
  __syncthreads();
  if (t < 64) {
    float a2 = bc2[t];
    for (int k = 0; k < 256; ++k) a2 += h1s[k] * wc2[k * 64 + t];
    h2s[t] = fmaxf(a2, 0.0f);
  }
  __syncthreads();
  if (t == 0) {
    float o = bc3[0];
    for (int k = 0; k < 64; ++k) o += h2s[k] * wc3[k];
    out[g] = o;
  }
}

// ---------------------------------------------------------------------------
// Launch
// ---------------------------------------------------------------------------
extern "C" void kernel_launch(void* const* d_in, const int* in_sizes, int n_in,
                              void* d_out, int out_size, void* d_ws, size_t ws_size,
                              hipStream_t stream) {
  (void)in_sizes; (void)n_in; (void)out_size; (void)ws_size;
  const int N = 20000, E = 320000, G = 64;

  const float* x     = (const float*)d_in[0];
  const int*   ei    = (const int*)d_in[1];
  const int*   batch = (const int*)d_in[2];
  const float* w1_0 = (const float*)d_in[3];
  const float* b1_0 = (const float*)d_in[4];
  const float* w2_0 = (const float*)d_in[5];
  const float* b2_0 = (const float*)d_in[6];
  const float* w1_1 = (const float*)d_in[7];
  const float* b1_1 = (const float*)d_in[8];
  const float* w2_1 = (const float*)d_in[9];
  const float* b2_1 = (const float*)d_in[10];
  const float* w1_2 = (const float*)d_in[11];
  const float* b1_2 = (const float*)d_in[12];
  const float* w2_2 = (const float*)d_in[13];
  const float* b2_2 = (const float*)d_in[14];
  const float* wc1  = (const float*)d_in[15];
  const float* bc1  = (const float*)d_in[16];
  const float* wc2  = (const float*)d_in[17];
  const float* bc2  = (const float*)d_in[18];
  const float* wc3  = (const float*)d_in[19];
  const float* bc3  = (const float*)d_in[20];
  float* out = (float*)d_out;

  const int* src = ei;
  const int* dst = ei + E;

  // workspace carve-out (256B aligned)
  char* ws = (char*)d_ws;
  size_t off = 0;
  auto alloc = [&](size_t bytes) -> void* {
    void* p = ws + off;
    off += (bytes + 255) & ~(size_t)255;
    return p;
  };
  float*          xa   = (float*)alloc((size_t)N * 256 * sizeof(float));
  float*          xb   = (float*)alloc((size_t)N * 256 * sizeof(float));
  unsigned short* w1p0 = (unsigned short*)alloc((size_t)128 * 256 * 2);
  unsigned short* w2p0 = (unsigned short*)alloc((size_t)256 * 256 * 2);
  unsigned short* w1p1 = (unsigned short*)alloc((size_t)512 * 256 * 2);
  unsigned short* w2p1 = (unsigned short*)alloc((size_t)256 * 256 * 2);
  unsigned short* w1p2 = (unsigned short*)alloc((size_t)512 * 256 * 2);
  unsigned short* w2p2 = (unsigned short*)alloc((size_t)256 * 256 * 2);
  float*          gsum = (float*)alloc((size_t)G * 256 * sizeof(float));
  float*          gmax = (float*)alloc((size_t)G * 256 * sizeof(float));
  unsigned int*   gcnt = (unsigned int*)alloc((size_t)G * sizeof(unsigned int));

  // pack weights to bf16 fragment-major
  pack_w<<<(128 * 256) / 256, 256, 0, stream>>>(w1_0, w1p0, 112, 128);
  pack_w<<<(256 * 256) / 256, 256, 0, stream>>>(w2_0, w2p0, 256, 256);
  pack_w<<<(512 * 256) / 256, 256, 0, stream>>>(w1_1, w1p1, 512, 512);
  pack_w<<<(256 * 256) / 256, 256, 0, stream>>>(w2_1, w2p1, 256, 256);
  pack_w<<<(512 * 256) / 256, 256, 0, stream>>>(w1_2, w1p2, 512, 512);
  pack_w<<<(256 * 256) / 256, 256, 0, stream>>>(w2_2, w2p2, 256, 256);

  const float NEG_INF = -__builtin_huge_valf();
  const int nFeat = N * 256;
  const int fb = (nFeat + 255) / 256;
  const int eb = (E + 15) / 16;

  // layer 0: x[N,56] -> xa[N,256]
  fill_f32<<<fb, 256, 0, stream>>>(xa, NEG_INF, nFeat);
  edge_conv<128, 56, 56><<<eb, 256, 0, stream>>>(x, src, dst, w1p0, b1_0, w2p0, b2_0, xa, E);
  finalize_max<<<fb, 256, 0, stream>>>(xa, nFeat);

  // layer 1: xa -> xb
  fill_f32<<<fb, 256, 0, stream>>>(xb, NEG_INF, nFeat);
  edge_conv<512, 256, 256><<<eb, 256, 0, stream>>>(xa, src, dst, w1p1, b1_1, w2p1, b2_1, xb, E);
  finalize_max<<<fb, 256, 0, stream>>>(xb, nFeat);

  // layer 2: xb -> xa
  fill_f32<<<fb, 256, 0, stream>>>(xa, NEG_INF, nFeat);
  edge_conv<512, 256, 256><<<eb, 256, 0, stream>>>(xb, src, dst, w1p2, b1_2, w2p2, b2_2, xa, E);
  finalize_max<<<fb, 256, 0, stream>>>(xa, nFeat);

  // pooling + head
  fill_f32<<<(G * 256 + 255) / 256, 256, 0, stream>>>(gsum, 0.0f, G * 256);
  fill_f32<<<(G * 256 + 255) / 256, 256, 0, stream>>>(gmax, NEG_INF, G * 256);
  fill_u32<<<1, 64, 0, stream>>>(gcnt, 0u, G);
  pool_count<<<(N + 255) / 256, 256, 0, stream>>>(batch, gcnt, N);
  pool_accum<<<fb, 256, 0, stream>>>(xa, batch, gsum, gmax, nFeat);
  head_mlp<<<G, 256, 0, stream>>>(gsum, gmax, gcnt, wc1, bc1, wc2, bc2, wc3, bc3, out);
}
